// ImprovedGraphAutoEncoder_3599182594602
// MI455X (gfx1250) — matrix operand order repaired and checked by reading, not compiled, not run on hardware
//
#include <hip/hip_runtime.h>
#include <hip/hip_bf16.h>
#include <math.h>

// ---------------------------------------------------------------------------
// CDNA5 / gfx1250 WMMA helpers (wave32).  D = A(16x32 f16) x B(32x16 f16) + C(f32)
// Fragment layouts per CDNA5 ISA 7.12.2.  Fragments are fed with 128-bit LDS
// loads: A tiles are row-major, B tiles are stored N-major (transposed) so each
// lane's fragment is two contiguous 16-byte runs.
// ---------------------------------------------------------------------------
typedef __attribute__((ext_vector_type(16))) _Float16 v16h;
typedef __attribute__((ext_vector_type(8)))  _Float16 v8h;
typedef __attribute__((ext_vector_type(8)))  float    v8f;

static constexpr int   NB  = 16;     // batches
static constexpr int   NP  = 256;    // points per batch
static constexpr int   ROWS = NB * NP;
static constexpr float NEG_SLOPE = 0.2f;
static constexpr float ALPHA_MIX = 0.1f;

union FragU { v16h v; v8h h[2]; };

// A fragment: rows 0..15 of row-major LDS tile (lda halves), K window [k0,k0+32).
// ISA layout: element e -> k = e + 8*half + (e&8 ? 8 : 0)  => two contiguous
// 8-half runs at k0+8*half and k0+16+8*half.
__device__ inline v16h frag_a(const _Float16* A, int lda, int k0, int lane) {
  const int h = lane >> 4, m = lane & 15;
  const _Float16* p = A + m * lda + k0 + 8 * h;
  FragU u;
  u.h[0] = *(const v8h*)(p);        // ds_load_b128
  u.h[1] = *(const v8h*)(p + 16);   // ds_load_b128
  return u.v;
}
// B fragment from N-major (transposed) LDS tile Bt[n][k] (ldk halves),
// K window [k0,k0+32), N window [n0,n0+16).
// ISA layout: element e -> k = e + 8*half + (e&8 ? 8 : 0)  => two contiguous
// 8-half runs at k0+8*half and k0+16+8*half.
__device__ inline v16h frag_bT(const _Float16* Bt, int ldk, int k0, int n0, int lane) {
  const int h = lane >> 4, n = lane & 15;
  const _Float16* p = Bt + (n0 + n) * ldk + k0 + 8 * h;
  FragU u;
  u.h[0] = *(const v8h*)(p);        // ds_load_b128
  u.h[1] = *(const v8h*)(p + 16);   // ds_load_b128
  return u.v;
}
__device__ inline v8f wmma32(v16h a, v16h b, v8f c) {
  // (neg_a, A, neg_b, B, c_mod, C, reuse_a, reuse_b)
  return __builtin_amdgcn_wmma_f32_16x16x32_f16(false, a, false, b, (short)0, c,
                                                false, false);
}

// ---------------------------------------------------------------------------
// Stage 1: encoder MLP  12 -> 64 -> 64 -> 3   (one wave per 16 rows)
// ---------------------------------------------------------------------------
__global__ __launch_bounds__(32) void enc_kernel(
    const float* __restrict__ x,
    const float* __restrict__ w1, const float* __restrict__ b1,
    const float* __restrict__ w2, const float* __restrict__ b2,
    const float* __restrict__ w3, const float* __restrict__ b3,
    float* __restrict__ lat, float* __restrict__ lat_out)
{
  __shared__ alignas(16) _Float16 sA[16 * 64];   // activations, row-major
  __shared__ alignas(16) _Float16 sWt[64 * 64];  // weights, N-major [n][k]
  __shared__ alignas(16) _Float16 sH[16 * 64];
  const int lane = threadIdx.x;
  const int row0 = blockIdx.x * 16;
  const int h = lane >> 4, nl = lane & 15;

  for (int idx = lane; idx < 16 * 64; idx += 32) {
    const int m = idx >> 6, k = idx & 63;
    sA[idx] = (_Float16)(k < 12 ? x[(row0 + m) * 12 + k] : 0.0f);
  }
  // W1 [12][64] -> sWt[n][k], K padded 12->64
  for (int idx = lane; idx < 64 * 64; idx += 32) {
    const int n = idx >> 6, k = idx & 63;
    sWt[idx] = (_Float16)(k < 12 ? w1[k * 64 + n] : 0.0f);
  }
  __syncthreads();
  // h1 = relu(x @ W1 + b1)   (K padded 12->32, one WMMA per 16-col tile)
  for (int nt = 0; nt < 4; ++nt) {
    v8f c = {};
    c = wmma32(frag_a(sA, 64, 0, lane), frag_bT(sWt, 64, 0, nt * 16, lane), c);
#pragma unroll
    for (int r = 0; r < 8; ++r) {
      const int m = r + 8 * h, n = nt * 16 + nl;
      sH[m * 64 + n] = (_Float16)fmaxf(c[r] + b1[n], 0.0f);
    }
  }
  __syncthreads();
  for (int idx = lane; idx < 64 * 64; idx += 32) {
    const int n = idx >> 6, k = idx & 63;
    sWt[idx] = (_Float16)w2[k * 64 + n];
  }
  __syncthreads();
  // h2 = relu(h1 @ W2 + b2)
  for (int nt = 0; nt < 4; ++nt) {
    v8f c = {};
    for (int kt = 0; kt < 2; ++kt)
      c = wmma32(frag_a(sH, 64, kt * 32, lane), frag_bT(sWt, 64, kt * 32, nt * 16, lane), c);
#pragma unroll
    for (int r = 0; r < 8; ++r) {
      const int m = r + 8 * h, n = nt * 16 + nl;
      sA[m * 64 + n] = (_Float16)fmaxf(c[r] + b2[n], 0.0f);   // reuse sA as h2
    }
  }
  __syncthreads();
  // W3 [64][3] -> sWt[n][k], N padded 3->16
  for (int idx = lane; idx < 16 * 64; idx += 32) {
    const int n = idx >> 6, k = idx & 63;
    sWt[idx] = (_Float16)(n < 3 ? w3[k * 3 + n] : 0.0f);
  }
  __syncthreads();
  {
    v8f c = {};
    for (int kt = 0; kt < 2; ++kt)
      c = wmma32(frag_a(sA, 64, kt * 32, lane), frag_bT(sWt, 64, kt * 32, 0, lane), c);
#pragma unroll
    for (int r = 0; r < 8; ++r) {
      const int m = r + 8 * h, n = nl;
      if (n < 3) {
        const float v = c[r] + b3[n];
        lat[(row0 + m) * 3 + n] = v;
        lat_out[(row0 + m) * 3 + n] = v;
      }
    }
  }
}

// ---------------------------------------------------------------------------
// Stage 2: Gabriel graph mask.  block = (batch, i); thread = j; min over k.
// ---------------------------------------------------------------------------
__global__ __launch_bounds__(256) void gabriel_kernel(
    const float* __restrict__ lat, unsigned char* __restrict__ mask)
{
  __shared__ float sp[NP * 3];
  const int b = blockIdx.x >> 8;
  const int i = blockIdx.x & 255;
  const int j = threadIdx.x;
  for (int t = threadIdx.x; t < NP * 3; t += 256) sp[t] = lat[b * NP * 3 + t];
  __syncthreads();
  const float pix = sp[i * 3], piy = sp[i * 3 + 1], piz = sp[i * 3 + 2];
  const float pjx = sp[j * 3], pjy = sp[j * 3 + 1], pjz = sp[j * 3 + 2];
  const float mx = 0.5f * (pix + pjx), my = 0.5f * (piy + pjy), mz = 0.5f * (piz + pjz);
  const float dx = pix - pjx, dy = piy - pjy, dz = piz - pjz;
  const float rad2 = 0.25f * (dx * dx + dy * dy + dz * dz);
  float minv = INFINITY;
  for (int k = 0; k < NP; ++k) {
    const float ex = sp[k * 3] - mx, ey = sp[k * 3 + 1] - my, ez = sp[k * 3 + 2] - mz;
    const float d2 = ex * ex + ey * ey + ez * ez;
    const bool excl = (k == i) || (k == j);
    minv = fminf(minv, excl ? INFINITY : d2);
  }
  mask[(b * NP + i) * NP + j] = ((minv >= rad2) && (i != j)) ? 1 : 0;
}

// ---------------------------------------------------------------------------
// Stage 3: self-loop attribute = mean neighbor distance (PyG fill_value='mean')
// ---------------------------------------------------------------------------
__global__ __launch_bounds__(256) void selfattr_kernel(
    const float* __restrict__ lat, const unsigned char* __restrict__ mask,
    float* __restrict__ self_attr)
{
  __shared__ float sp[NP * 3];
  const int b = blockIdx.x, i = threadIdx.x;
  for (int t = i; t < NP * 3; t += 256) sp[t] = lat[b * NP * 3 + t];
  __syncthreads();
  const float pix = sp[i * 3], piy = sp[i * 3 + 1], piz = sp[i * 3 + 2];
  float deg = 0.0f, s = 0.0f;
  const unsigned char* mrow = mask + (b * NP + i) * NP;
  for (int j = 0; j < NP; ++j) {
    if (mrow[j]) {
      const float dx = pix - sp[j * 3], dy = piy - sp[j * 3 + 1], dz = piz - sp[j * 3 + 2];
      const float d2 = dx * dx + dy * dy + dz * dz;
      deg += 1.0f;
      s += (d2 > 0.0f) ? sqrtf(d2) : 0.0f;
    }
  }
  self_attr[b * NP + i] = (deg > 0.0f) ? s / deg : 0.0f;
}

// ---------------------------------------------------------------------------
// hl/hr projections.  Layer 1 (Fin=1) scalar; layers 2/3 (Fin=64) via WMMA.
// ---------------------------------------------------------------------------
__global__ void g1_hlhr_kernel(const float* __restrict__ lat,
                               const float* __restrict__ wl, const float* __restrict__ bl,
                               const float* __restrict__ wr, const float* __restrict__ br,
                               float* __restrict__ hl, float* __restrict__ hr)
{
  const int idx = blockIdx.x * blockDim.x + threadIdx.x;
  if (idx >= ROWS * 64) return;
  const int row = idx >> 6, f = idx & 63;
  const float x = lat[row * 3 + 2];                 // x = lat[..., 2:3]
  hl[idx] = x * wl[f] + bl[f];
  hr[idx] = x * wr[f] + br[f];
}

template <int FOUT>
__global__ __launch_bounds__(32) void hlhr_wmma_kernel(
    const float* __restrict__ x,
    const float* __restrict__ wl, const float* __restrict__ bl,
    const float* __restrict__ wr, const float* __restrict__ br,
    float* __restrict__ hl, float* __restrict__ hr)
{
  constexpr int FP = (FOUT + 15) & ~15;             // 64 or 16
  __shared__ alignas(16) _Float16 sA[16 * 64];
  __shared__ alignas(16) _Float16 sWt[FP * 64];     // N-major [n][k]
  const int lane = threadIdx.x, row0 = blockIdx.x * 16;
  const int h = lane >> 4, nl = lane & 15;

  for (int idx = lane; idx < 16 * 64; idx += 32)
    sA[idx] = (_Float16)x[row0 * 64 + idx];

  for (int pass = 0; pass < 2; ++pass) {
    const float* w  = pass ? wr : wl;
    const float* bb = pass ? br : bl;
    float* out      = pass ? hr : hl;
    __syncthreads();
    for (int idx = lane; idx < FP * 64; idx += 32) {
      const int n = idx >> 6, k = idx & 63;
      sWt[idx] = (_Float16)(n < FOUT ? w[k * FOUT + n] : 0.0f);
    }
    __syncthreads();
    for (int nt = 0; nt < FP / 16; ++nt) {
      v8f c = {};
      for (int kt = 0; kt < 2; ++kt)
        c = wmma32(frag_a(sA, 64, kt * 32, lane), frag_bT(sWt, 64, kt * 32, nt * 16, lane), c);
#pragma unroll
      for (int r = 0; r < 8; ++r) {
        const int m = r + 8 * h, n = nt * 16 + nl;
        if (n < FOUT) out[(row0 + m) * FOUT + n] = c[r] + bb[n];
      }
    }
  }
}

// ---------------------------------------------------------------------------
// GATv2 attention: scores + masked softmax (VALU, leakyrelu is elementwise
// nonlinear), then alpha @ hl via WMMA (K=256 in 8 chunks of 32).
// Block = (batch, 16-row tile), 256 threads = 8 waves.
// hl is staged in LDS transposed (N-major) so it serves both the score loop
// and the WMMA B fragments (128-bit ds loads).
// ---------------------------------------------------------------------------
template <int FOUT, bool RELU>
__global__ __launch_bounds__(256) void attn_kernel(
    const float* __restrict__ hl, const float* __restrict__ hr,
    const unsigned char* __restrict__ mask, const float* __restrict__ lat,
    const float* __restrict__ self_attr, const float* __restrict__ we,
    const float* __restrict__ att, const float* __restrict__ bias,
    float* __restrict__ out)
{
  constexpr int FP = (FOUT + 15) & ~15;
  __shared__ alignas(16) _Float16 s_hlT[FP * NP];   // [n][j]  32 KB / 8 KB
  __shared__ float    s_sc[16 * NP];                // 16 KB scores
  __shared__ alignas(16) _Float16 s_al[16 * NP];    // 8 KB alpha (A fragments)
  __shared__ float    s_lat[NP * 3];
  __shared__ float    s_red[16 * 16];
  __shared__ float    s_we[FOUT], s_att[FOUT];

  const int t = threadIdx.x;
  const int b = blockIdx.x >> 4;
  const int tile = blockIdx.x & 15;
  const int i_loc = t >> 4, jl = t & 15;
  const int i = tile * 16 + i_loc;

  for (int idx = t; idx < NP * FP; idx += 256) {
    const int j = idx / FP, n = idx % FP;           // coalesced global read
    s_hlT[n * NP + j] = (_Float16)(n < FOUT ? hl[(b * NP + j) * FOUT + n] : 0.0f);
  }
  for (int idx = t; idx < NP * 3; idx += 256) s_lat[idx] = lat[b * NP * 3 + idx];
  if (t < FOUT) { s_we[t] = we[t]; s_att[t] = att[t]; }
  __builtin_prefetch(mask + (b * NP + i) * NP, 0, 0);   // global_prefetch_b8
  __syncthreads();

  // ---- scores: thread owns row i, 16 of the 256 j columns -----------------
  const float pix = s_lat[i * 3], piy = s_lat[i * 3 + 1], piz = s_lat[i * 3 + 2];
  const float sa_i = self_attr[b * NP + i];
  const unsigned char* mrow = mask + (b * NP + i) * NP;
  const float* hri = hr + (b * NP + i) * FOUT;
  for (int jj = 0; jj < 16; ++jj) {
    const int j = jl + jj * 16;
    const bool allowed = (j == i) || mrow[j];
    float s;
    if (!allowed) {
      s = -INFINITY;
    } else {
      float e;
      if (j == i) {
        e = sa_i;
      } else {
        const float dx = pix - s_lat[j * 3], dy = piy - s_lat[j * 3 + 1],
                    dz = piz - s_lat[j * 3 + 2];
        const float d2 = dx * dx + dy * dy + dz * dz;
        e = (d2 > 0.0f) ? sqrtf(d2) : 0.0f;
      }
      s = 0.0f;
      for (int c = 0; c < FOUT; ++c) {
        float z = hri[c] + (float)s_hlT[c * NP + j] + e * s_we[c];
        z = (z > 0.0f) ? z : z * NEG_SLOPE;
        s += s_att[c] * z;
      }
    }
    s_sc[i_loc * NP + j] = s;
  }
  __syncthreads();

  // ---- softmax over j (16 threads cooperate per row) ----------------------
  float lmax = -INFINITY;
  for (int jj = 0; jj < 16; ++jj) lmax = fmaxf(lmax, s_sc[i_loc * NP + jl + jj * 16]);
  s_red[i_loc * 16 + jl] = lmax;
  __syncthreads();
  float rmax = -INFINITY;
  for (int q = 0; q < 16; ++q) rmax = fmaxf(rmax, s_red[i_loc * 16 + q]);
  float lsum = 0.0f;
  for (int jj = 0; jj < 16; ++jj) lsum += expf(s_sc[i_loc * NP + jl + jj * 16] - rmax);
  __syncthreads();
  s_red[i_loc * 16 + jl] = lsum;
  __syncthreads();
  float rsum = 0.0f;
  for (int q = 0; q < 16; ++q) rsum += s_red[i_loc * 16 + q];
  const float rinv = 1.0f / rsum;
  for (int jj = 0; jj < 16; ++jj) {
    const int j = jl + jj * 16;
    s_al[i_loc * NP + j] = (_Float16)(expf(s_sc[i_loc * NP + j] - rmax) * rinv);
  }
  __syncthreads();

  // ---- alpha @ hl via WMMA (wave-uniform branch, EXEC all-ones) -----------
  const int wave = t >> 5, lane = t & 31;
  if (wave < FP / 16) {
    v8f c = {};
    for (int kt = 0; kt < 8; ++kt)
      c = wmma32(frag_a(s_al, NP, kt * 32, lane),
                 frag_bT(s_hlT, NP, kt * 32, wave * 16, lane), c);
    const int h = lane >> 4, nl2 = lane & 15;
#pragma unroll
    for (int r = 0; r < 8; ++r) {
      const int m = r + 8 * h, n = wave * 16 + nl2;
      if (n < FOUT) {
        float v = c[r] + bias[n];
        if (RELU) v = fmaxf(v, 0.0f);
        out[(b * NP + tile * 16 + m) * FOUT + n] = v;
      }
    }
  }
}

// ---------------------------------------------------------------------------
// Stage 4: combined = x3 + 0.1*skip(lat); recon = head(combined). All WMMA.
// ---------------------------------------------------------------------------
__global__ __launch_bounds__(32) void tail_kernel(
    const float* __restrict__ x3, const float* __restrict__ lat,
    const float* s1w, const float* s1b, const float* s2w, const float* s2b,
    const float* s3w, const float* s3b,
    const float* h1w, const float* h1b, const float* h2w, const float* h2b,
    const float* h3w, const float* h3b,
    float* __restrict__ recon)
{
  __shared__ alignas(16) _Float16 sA[16 * 64];
  __shared__ alignas(16) _Float16 sWt[64 * 64];     // N-major [n][k]
  __shared__ alignas(16) _Float16 sH[16 * 64];
  __shared__ float    sComb[16 * 3];
  const int lane = threadIdx.x, row0 = blockIdx.x * 16;
  const int h = lane >> 4, nl = lane & 15;

  // ---- skip: lat(3) -> 64 -> 64 -> 3 --------------------------------------
  for (int idx = lane; idx < 16 * 64; idx += 32) {
    const int m = idx >> 6, k = idx & 63;
    sA[idx] = (_Float16)(k < 3 ? lat[(row0 + m) * 3 + k] : 0.0f);
  }
  for (int idx = lane; idx < 64 * 64; idx += 32) {
    const int n = idx >> 6, k = idx & 63;
    sWt[idx] = (_Float16)(k < 3 ? s1w[k * 64 + n] : 0.0f);
  }
  __syncthreads();
  for (int nt = 0; nt < 4; ++nt) {
    v8f c = {};
    c = wmma32(frag_a(sA, 64, 0, lane), frag_bT(sWt, 64, 0, nt * 16, lane), c);
#pragma unroll
    for (int r = 0; r < 8; ++r) {
      const int m = r + 8 * h, n = nt * 16 + nl;
      sH[m * 64 + n] = (_Float16)fmaxf(c[r] + s1b[n], 0.0f);
    }
  }
  __syncthreads();
  for (int idx = lane; idx < 64 * 64; idx += 32) {
    const int n = idx >> 6, k = idx & 63;
    sWt[idx] = (_Float16)s2w[k * 64 + n];
  }
  __syncthreads();
  for (int nt = 0; nt < 4; ++nt) {
    v8f c = {};
    for (int kt = 0; kt < 2; ++kt)
      c = wmma32(frag_a(sH, 64, kt * 32, lane), frag_bT(sWt, 64, kt * 32, nt * 16, lane), c);
#pragma unroll
    for (int r = 0; r < 8; ++r) {
      const int m = r + 8 * h, n = nt * 16 + nl;
      sA[m * 64 + n] = (_Float16)fmaxf(c[r] + s2b[n], 0.0f);
    }
  }
  __syncthreads();
  for (int idx = lane; idx < 16 * 64; idx += 32) {
    const int n = idx >> 6, k = idx & 63;
    sWt[idx] = (_Float16)(n < 3 ? s3w[k * 3 + n] : 0.0f);
  }
  __syncthreads();
  {
    v8f c = {};
    for (int kt = 0; kt < 2; ++kt)
      c = wmma32(frag_a(sA, 64, kt * 32, lane), frag_bT(sWt, 64, kt * 32, 0, lane), c);
#pragma unroll
    for (int r = 0; r < 8; ++r) {
      const int m = r + 8 * h, n = nl;
      if (n < 3)
        sComb[m * 3 + n] = x3[(row0 + m) * 3 + n] + ALPHA_MIX * (c[r] + s3b[n]);
    }
  }
  __syncthreads();

  // ---- head: combined(3) -> 64 -> 64 -> 12 --------------------------------
  for (int idx = lane; idx < 16 * 64; idx += 32) {
    const int m = idx >> 6, k = idx & 63;
    sA[idx] = (_Float16)(k < 3 ? sComb[m * 3 + k] : 0.0f);
  }
  for (int idx = lane; idx < 64 * 64; idx += 32) {
    const int n = idx >> 6, k = idx & 63;
    sWt[idx] = (_Float16)(k < 3 ? h1w[k * 64 + n] : 0.0f);
  }
  __syncthreads();
  for (int nt = 0; nt < 4; ++nt) {
    v8f c = {};
    c = wmma32(frag_a(sA, 64, 0, lane), frag_bT(sWt, 64, 0, nt * 16, lane), c);
#pragma unroll
    for (int r = 0; r < 8; ++r) {
      const int m = r + 8 * h, n = nt * 16 + nl;
      sH[m * 64 + n] = (_Float16)fmaxf(c[r] + h1b[n], 0.0f);
    }
  }
  __syncthreads();
  for (int idx = lane; idx < 64 * 64; idx += 32) {
    const int n = idx >> 6, k = idx & 63;
    sWt[idx] = (_Float16)h2w[k * 64 + n];
  }
  __syncthreads();
  for (int nt = 0; nt < 4; ++nt) {
    v8f c = {};
    for (int kt = 0; kt < 2; ++kt)
      c = wmma32(frag_a(sH, 64, kt * 32, lane), frag_bT(sWt, 64, kt * 32, nt * 16, lane), c);
#pragma unroll
    for (int r = 0; r < 8; ++r) {
      const int m = r + 8 * h, n = nt * 16 + nl;
      sA[m * 64 + n] = (_Float16)fmaxf(c[r] + h2b[n], 0.0f);
    }
  }
  __syncthreads();
  for (int idx = lane; idx < 16 * 64; idx += 32) {
    const int n = idx >> 6, k = idx & 63;
    sWt[idx] = (_Float16)(n < 12 ? h3w[k * 12 + n] : 0.0f);
  }
  __syncthreads();
  {
    v8f c = {};
    for (int kt = 0; kt < 2; ++kt)
      c = wmma32(frag_a(sA, 64, kt * 32, lane), frag_bT(sWt, 64, kt * 32, 0, lane), c);
#pragma unroll
    for (int r = 0; r < 8; ++r) {
      const int m = r + 8 * h, n = nl;
      if (n < 12) recon[(row0 + m) * 12 + n] = c[r] + h3b[n];
    }
  }
}

// ---------------------------------------------------------------------------
// Host-side pipeline
// ---------------------------------------------------------------------------
extern "C" void kernel_launch(void* const* d_in, const int* in_sizes, int n_in,
                              void* d_out, int out_size, void* d_ws, size_t ws_size,
                              hipStream_t stream)
{
  (void)in_sizes; (void)n_in; (void)out_size; (void)ws_size;
  const float* batch = (const float*)d_in[0];
  auto f = [&](int i) { return (const float*)d_in[i]; };

  char* ws = (char*)d_ws;
  float*         lat       = (float*)(ws + 0);          // 16*256*3 f
  unsigned char* mask      = (unsigned char*)(ws + 49152);   // 16*256*256 u8
  float*         self_attr = (float*)(ws + 1097728);    // 16*256 f
  float*         hl        = (float*)(ws + 1114112);    // 16*256*64 f
  float*         hr        = (float*)(ws + 2162688);    // 16*256*64 f
  float*         xa        = (float*)(ws + 3211264);    // 16*256*64 f
  float*         xb        = (float*)(ws + 4259840);    // 16*256*64 f
  float*         x3        = (float*)(ws + 5308416);    // 16*256*3 f

  float* recon   = (float*)d_out;                       // 16*256*12
  float* lat_out = recon + NB * NP * 12;                // 16*256*3

  // encoder -> lat (ws + output)
  enc_kernel<<<ROWS / 16, 32, 0, stream>>>(batch, f(1), f(2), f(3), f(4), f(5), f(6),
                                           lat, lat_out);
  // Gabriel graph + self-loop attr
  gabriel_kernel<<<NB * NP, 256, 0, stream>>>(lat, mask);
  selfattr_kernel<<<NB, 256, 0, stream>>>(lat, mask, self_attr);

  // GAT layer 1: Fin=1, Fout=64
  g1_hlhr_kernel<<<(ROWS * 64) / 256, 256, 0, stream>>>(lat, f(7), f(8), f(9), f(10), hl, hr);
  attn_kernel<64, true><<<NB * 16, 256, 0, stream>>>(hl, hr, mask, lat, self_attr,
                                                     f(11), f(12), f(13), xa);
  // GAT layer 2: 64 -> 64
  hlhr_wmma_kernel<64><<<ROWS / 16, 32, 0, stream>>>(xa, f(14), f(15), f(16), f(17), hl, hr);
  attn_kernel<64, true><<<NB * 16, 256, 0, stream>>>(hl, hr, mask, lat, self_attr,
                                                     f(18), f(19), f(20), xb);
  // GAT layer 3: 64 -> 3 (no relu)
  hlhr_wmma_kernel<3><<<ROWS / 16, 32, 0, stream>>>(xb, f(21), f(22), f(23), f(24), hl, hr);
  attn_kernel<3, false><<<NB * 16, 256, 0, stream>>>(hl, hr, mask, lat, self_attr,
                                                     f(25), f(26), f(27), x3);
  // skip + head -> recon
  tail_kernel<<<ROWS / 16, 32, 0, stream>>>(x3, lat,
                                            f(28), f(29), f(30), f(31), f(32), f(33),
                                            f(34), f(35), f(36), f(37), f(38), f(39),
                                            recon);
}
